// MultiHeadAttention_35175782154807
// MI455X (gfx1250) — matrix operand order repaired and checked by reading, not compiled, not run on hardware
//
#include <hip/hip_runtime.h>
#include <hip/hip_bf16.h>
#include <math.h>

typedef __bf16 bf16_t;
typedef bf16_t v16bf __attribute__((ext_vector_type(16)));
typedef float  v8f   __attribute__((ext_vector_type(8)));
typedef unsigned int u32x4 __attribute__((ext_vector_type(4)));
typedef int          i32x4 __attribute__((ext_vector_type(4)));
typedef int          i32x8 __attribute__((ext_vector_type(8)));

#define B_  2
#define T_  2048
#define C_  1024
#define H_  16
#define HD_ 64

__device__ __forceinline__ int lane_id() { return threadIdx.x & 31; }

// A fragment (16x32 bf16), A row-major with leading dim lda, tile origin (m0,k0).
// ISA layout: lane l: m = l&15, g = l>>4; VGPR j<4: k = 8g+2j(+1); j>=4: k = 16+8g+2(j-4)(+1)
__device__ __forceinline__ v16bf load_a_frag(const bf16_t* A, int lda, int m0, int k0) {
  int l = lane_id();
  int m = m0 + (l & 15);
  int g = l >> 4;
  const bf16_t* row = A + (size_t)m * lda + k0;
  v16bf a;
#pragma unroll
  for (int j = 0; j < 8; ++j) {
    int k = (j < 4) ? (8 * g + 2 * j) : (16 + 8 * g + 2 * (j - 4));
    a[2 * j]     = row[k];
    a[2 * j + 1] = row[k + 1];
  }
  return a;
}

// B fragment (32x16 bf16) from BT stored row-major [N][ldb] (BT[n][k] == B[k][n]).
// ISA layout: lane l: n = l&15, g = l>>4; VGPR j<4: k = 8g+2j(+1); j>=4: k = 16+8g+2(j-4)(+1)
__device__ __forceinline__ v16bf load_b_frag(const bf16_t* BT, int ldb, int n0, int k0) {
  int l = lane_id();
  int n = n0 + (l & 15);
  int g = l >> 4;
  const bf16_t* row = BT + (size_t)n * ldb + k0;
  v16bf b;
#pragma unroll
  for (int j = 0; j < 8; ++j) {
    int k = (j < 4) ? (8 * g + 2 * j) : (16 + 8 * g + 2 * (j - 4));
    b[2 * j]     = row[k];
    b[2 * j + 1] = row[k + 1];
  }
  return b;
}

__device__ __forceinline__ v8f wmma_bf16(v16bf a, v16bf b, v8f c) {
  return __builtin_amdgcn_wmma_f32_16x16x32_bf16(false, a, false, b, (short)0, c, false, false);
}

// ---- Tensor Data Mover: 2D tile (bf16) from global -> LDS, per CDNA5 D# layout ----
// group0: [1:0]=count=1, [63:32]=lds_addr, [120:64]=global_addr, [127:126]=type=2
// group1: [17:16]=data_size=1 (2B), [79:48]=tensor_dim0, [111:80]=tensor_dim1,
//         [127:112]=tile_dim0, [143:128]=tile_dim1, [207:160]=tensor_dim0_stride
__device__ __forceinline__ void tdm_load_2d(unsigned int lds_off, const void* gaddr,
                                            unsigned int dim0, unsigned int dim1,
                                            unsigned int tile0, unsigned int tile1,
                                            unsigned long long stride0_elems) {
  unsigned long long ga = (unsigned long long)(uintptr_t)gaddr;
  u32x4 g0;
  g0[0] = 1u;                                                   // count=1, user descriptor
  g0[1] = lds_off;                                              // LDS byte address
  g0[2] = (unsigned int)ga;                                     // global_addr[31:0]
  g0[3] = (unsigned int)((ga >> 32) & 0x1FFFFFFu) | (2u << 30); // global_addr[56:32] | type=2
  i32x8 g1;
  g1[0] = (int)(1u << 16);                                      // wg_mask=0, data_size=2B
  g1[1] = (int)((dim0 & 0xFFFFu) << 16);                        // tensor_dim0[15:0]
  g1[2] = (int)((dim0 >> 16) | ((dim1 & 0xFFFFu) << 16));       // dim0[31:16] | dim1[15:0]
  g1[3] = (int)((dim1 >> 16) | (tile0 << 16));                  // dim1[31:16] | tile_dim0
  g1[4] = (int)tile1;                                           // tile_dim1 (tile_dim2=0)
  g1[5] = (int)(unsigned int)(stride0_elems & 0xFFFFFFFFu);     // stride0[31:0]
  g1[6] = (int)(unsigned int)((stride0_elems >> 32) & 0xFFFFu); // stride0[47:32], stride1=0
  g1[7] = 0;
  i32x4 z4 = {0, 0, 0, 0};
#if __clang_major__ >= 23
  i32x8 z8 = {0, 0, 0, 0, 0, 0, 0, 0};
  __builtin_amdgcn_tensor_load_to_lds(g0, g1, z4, z4, z8, 0);
#else
  __builtin_amdgcn_tensor_load_to_lds(g0, g1, z4, z4, 0);
#endif
}

// ---------------- prep kernels ----------------

__global__ void cvt_f32_bf16(const float* __restrict__ src, bf16_t* __restrict__ dst, int n) {
  int i = blockIdx.x * blockDim.x + threadIdx.x;
  if (i < n) dst[i] = (bf16_t)src[i];
}

// W: [H][C][HD] f32 -> Wt: [H][HD][C] bf16 (so GEMM B-fragments read contiguous K pairs)
__global__ void transpose_w(const float* __restrict__ W, bf16_t* __restrict__ Wt) {
  int i = blockIdx.x * blockDim.x + threadIdx.x;
  if (i >= H_ * C_ * HD_) return;
  int h = i / (C_ * HD_);
  int r = i % (C_ * HD_);
  int c = r / HD_;
  int d = r % HD_;
  Wt[((size_t)h * HD_ + d) * C_ + c] = (bf16_t)W[i];
}

// ---------------- QKV projection: one wave computes a 32x64 tile of Q, K or V ----------------

__global__ void qkv_gemm(const bf16_t* __restrict__ xb,
                         const bf16_t* __restrict__ wqt,
                         const bf16_t* __restrict__ wkt,
                         const bf16_t* __restrict__ wvt,
                         bf16_t* __restrict__ q,
                         bf16_t* __restrict__ k,
                         bf16_t* __restrict__ vt) {
  int i0  = blockIdx.x * 32;
  int bh  = blockIdx.y;      // b*H + h
  int mat = blockIdx.z;      // 0=Q, 1=K, 2=V
  int b = bh / H_, h = bh % H_;
  const bf16_t* A  = xb + (size_t)b * T_ * C_;
  const bf16_t* Wt = (mat == 0 ? wqt : (mat == 1 ? wkt : wvt)) + (size_t)h * HD_ * C_;

  v8f acc[2][4] = {};
  for (int k0 = 0; k0 < C_; k0 += 32) {
    v16bf a0 = load_a_frag(A, C_, i0, k0);
    v16bf a1 = load_a_frag(A, C_, i0 + 16, k0);
#pragma unroll
    for (int nt = 0; nt < 4; ++nt) {
      v16bf bf = load_b_frag(Wt, C_, nt * 16, k0);
      acc[0][nt] = wmma_bf16(a0, bf, acc[0][nt]);
      acc[1][nt] = wmma_bf16(a1, bf, acc[1][nt]);
    }
  }

  int l = lane_id(), n = l & 15, g = l >> 4;
  if (mat < 2) {
    bf16_t* dst = (mat == 0 ? q : k) + (size_t)bh * T_ * HD_;
#pragma unroll
    for (int half = 0; half < 2; ++half)
#pragma unroll
      for (int nt = 0; nt < 4; ++nt)
#pragma unroll
        for (int r = 0; r < 8; ++r) {
          int m = i0 + half * 16 + r + 8 * g;
          dst[(size_t)m * HD_ + nt * 16 + n] = (bf16_t)acc[half][nt][r];
        }
  } else {
    // store V transposed: Vt[bh][d][t] so the PV B-operand reads contiguous K(=t) pairs
    bf16_t* dst = vt + (size_t)bh * HD_ * T_;
#pragma unroll
    for (int half = 0; half < 2; ++half)
#pragma unroll
      for (int nt = 0; nt < 4; ++nt)
#pragma unroll
        for (int r = 0; r < 8; ++r) {
          int m = i0 + half * 16 + r + 8 * g;
          int d = nt * 16 + n;
          dst[(size_t)d * T_ + m] = (bf16_t)acc[half][nt][r];
        }
  }
}

// ---------------- causal flash attention: one wave per (b,h, 16-row tile) ----------------
// K/V tiles are DMA'd into LDS by the Tensor Data Mover, double-buffered, synced with TENSORcnt.

__global__ void flash_attn(const bf16_t* __restrict__ q,
                           const bf16_t* __restrict__ kmat,
                           const bf16_t* __restrict__ vt,
                           bf16_t* __restrict__ ob) {
  __shared__ bf16_t kbuf[2][32 * 64];  // K tile: 32 keys x 64 dims (row-major)
  __shared__ bf16_t vbuf[2][64 * 32];  // V^T tile: 64 dims x 32 keys (row-major)
  __shared__ bf16_t pbuf[16 * 32];     // P staging for C/D-layout -> A-layout re-pack

  int i0 = blockIdx.x * 16;
  int bh = blockIdx.y;
  int b = bh / H_, h = bh % H_;
  const bf16_t* Q = q    + (size_t)bh * T_ * HD_;
  const bf16_t* K = kmat + (size_t)bh * T_ * HD_;
  const bf16_t* V = vt   + (size_t)bh * HD_ * T_;

  v16bf aq0 = load_a_frag(Q, HD_, i0, 0);
  v16bf aq1 = load_a_frag(Q, HD_, i0, 32);

  v8f o[4] = {};
  float rmax[8], rsum[8];
#pragma unroll
  for (int r = 0; r < 8; ++r) { rmax[r] = -INFINITY; rsum[r] = 0.f; }

  int l = lane_id(), n = l & 15, g = l >> 4;
  const float scale = 0.125f;  // 1/sqrt(HD)

  int jbmax = (i0 + 15) >> 5;

  // prefetch first K/V tile pair via TDM
  tdm_load_2d((unsigned int)(uintptr_t)&kbuf[0][0], K, HD_, T_, 64, 32, HD_);
  tdm_load_2d((unsigned int)(uintptr_t)&vbuf[0][0], V, T_, HD_, 32, 64, T_);

  for (int jb = 0; jb <= jbmax; ++jb) {
    int j0 = jb * 32;
    int cur = jb & 1;
    if (jb < jbmax) {  // async-prefetch next tile pair into the other buffer
      int nj0 = j0 + 32;
      tdm_load_2d((unsigned int)(uintptr_t)&kbuf[cur ^ 1][0], K + (size_t)nj0 * HD_, HD_, T_, 64, 32, HD_);
      tdm_load_2d((unsigned int)(uintptr_t)&vbuf[cur ^ 1][0], V + nj0, T_, HD_, 32, 64, T_);
      __builtin_amdgcn_s_wait_tensorcnt(2);  // current pair complete, next pair in flight
    } else {
      __builtin_amdgcn_s_wait_tensorcnt(0);
    }
    const bf16_t* kt = &kbuf[cur][0];
    const bf16_t* vtile = &vbuf[cur][0];

    // S block: rows i0..i0+15, cols j0..j0+31 (two 16x16 f32 tiles, K=64 in two WMMA steps)
    v8f s0 = {}, s1 = {};
    {
      v16bf bk;
      bk = load_b_frag(kt, HD_, 0,  0 ); s0 = wmma_bf16(aq0, bk, s0);
      bk = load_b_frag(kt, HD_, 0,  32); s0 = wmma_bf16(aq1, bk, s0);
      bk = load_b_frag(kt, HD_, 16, 0 ); s1 = wmma_bf16(aq0, bk, s1);
      bk = load_b_frag(kt, HD_, 16, 32); s1 = wmma_bf16(aq1, bk, s1);
    }
    // online softmax per row (rows live entirely within a 16-lane half-wave)
#pragma unroll
    for (int r = 0; r < 8; ++r) {
      int m = i0 + r + 8 * g;
      float e0 = (j0 + n      <= m) ? s0[r] * scale : -INFINITY;
      float e1 = (j0 + 16 + n <= m) ? s1[r] * scale : -INFINITY;
      float mx = fmaxf(e0, e1);
#pragma unroll
      for (int off = 1; off < 16; off <<= 1)
        mx = fmaxf(mx, __shfl_xor(mx, off, 32));
      float nm = fmaxf(rmax[r], mx);
      float sc = __expf(rmax[r] - nm);
      rmax[r] = nm;
      float p0 = __expf(e0 - nm);
      float p1 = __expf(e1 - nm);
      float sr = p0 + p1;
#pragma unroll
      for (int off = 1; off < 16; off <<= 1)
        sr += __shfl_xor(sr, off, 32);
      rsum[r] = rsum[r] * sc + sr;
#pragma unroll
      for (int nt = 0; nt < 4; ++nt) o[nt][r] *= sc;
      pbuf[(r + 8 * g) * 32 + n]      = (bf16_t)p0;
      pbuf[(r + 8 * g) * 32 + 16 + n] = (bf16_t)p1;
    }
    __syncthreads();                       // single-wave block: just forces dscnt ordering
    v16bf ap = load_a_frag(pbuf, 32, 0, 0);
    __syncthreads();
#pragma unroll
    for (int nt = 0; nt < 4; ++nt) {
      v16bf bv = load_b_frag(vtile, 32, nt * 16, 0);
      o[nt] = wmma_bf16(ap, bv, o[nt]);
    }
  }

  // normalize + store bf16 in concat-head layout [B,T,C]
#pragma unroll
  for (int nt = 0; nt < 4; ++nt)
#pragma unroll
    for (int r = 0; r < 8; ++r) {
      int m = i0 + r + 8 * g;
      float val = o[nt][r] / rsum[r];
      ob[((size_t)b * T_ + m) * C_ + h * HD_ + nt * 16 + n] = (bf16_t)val;
    }
}

// ---------------- output projection: y = ob @ Wo^T + bo (one wave per 32x64 tile) ----------------

__global__ void out_proj(const bf16_t* __restrict__ ob,
                         const bf16_t* __restrict__ wob,
                         const float* __restrict__ bo,
                         float* __restrict__ y) {
  int m0 = blockIdx.x * 32;
  int n0 = blockIdx.y * 64;
  v8f acc[2][4] = {};
  for (int k0 = 0; k0 < C_; k0 += 32) {
    v16bf a0 = load_a_frag(ob, C_, m0, k0);
    v16bf a1 = load_a_frag(ob, C_, m0 + 16, k0);
#pragma unroll
    for (int nt = 0; nt < 4; ++nt) {
      v16bf bf = load_b_frag(wob, C_, n0 + nt * 16, k0);
      acc[0][nt] = wmma_bf16(a0, bf, acc[0][nt]);
      acc[1][nt] = wmma_bf16(a1, bf, acc[1][nt]);
    }
  }
  int l = lane_id(), n = l & 15, g = l >> 4;
#pragma unroll
  for (int half = 0; half < 2; ++half)
#pragma unroll
    for (int nt = 0; nt < 4; ++nt) {
      int col = n0 + nt * 16 + n;
      float bias = bo[col];
#pragma unroll
      for (int r = 0; r < 8; ++r) {
        int m = m0 + half * 16 + r + 8 * g;
        y[(size_t)m * C_ + col] = acc[half][nt][r] + bias;
      }
    }
}

// ---------------- host launcher ----------------

extern "C" void kernel_launch(void* const* d_in, const int* in_sizes, int n_in,
                              void* d_out, int out_size, void* d_ws, size_t ws_size,
                              hipStream_t stream) {
  const float* x  = (const float*)d_in[0];
  const float* Wq = (const float*)d_in[1];
  const float* Wk = (const float*)d_in[2];
  const float* Wv = (const float*)d_in[3];
  const float* Wo = (const float*)d_in[4];
  const float* bo = (const float*)d_in[5];
  float* y = (float*)d_out;

  char* ws = (char*)d_ws;
  size_t off = 0;
  auto alloc = [&](size_t bytes) -> char* {
    char* p = ws + off;
    off += (bytes + 255) & ~(size_t)255;
    return p;
  };
  bf16_t* xb  = (bf16_t*)alloc((size_t)B_ * T_ * C_ * 2);        // x in bf16
  bf16_t* wqt = (bf16_t*)alloc((size_t)H_ * HD_ * C_ * 2);       // Wq^T
  bf16_t* wkt = (bf16_t*)alloc((size_t)H_ * HD_ * C_ * 2);       // Wk^T
  bf16_t* wvt = (bf16_t*)alloc((size_t)H_ * HD_ * C_ * 2);       // Wv^T
  bf16_t* wob = (bf16_t*)alloc((size_t)C_ * C_ * 2);             // Wo bf16
  bf16_t* qb  = (bf16_t*)alloc((size_t)B_ * H_ * T_ * HD_ * 2);  // Q [B,H,T,HD]
  bf16_t* kb  = (bf16_t*)alloc((size_t)B_ * H_ * T_ * HD_ * 2);  // K [B,H,T,HD]
  bf16_t* vtb = (bf16_t*)alloc((size_t)B_ * H_ * T_ * HD_ * 2);  // V^T [B,H,HD,T]
  bf16_t* obb = (bf16_t*)alloc((size_t)B_ * T_ * C_ * 2);        // attention out bf16

  {
    int nX = B_ * T_ * C_;
    cvt_f32_bf16<<<(nX + 255) / 256, 256, 0, stream>>>(x, xb, nX);
    int nW = H_ * C_ * HD_;
    transpose_w<<<(nW + 255) / 256, 256, 0, stream>>>(Wq, wqt);
    transpose_w<<<(nW + 255) / 256, 256, 0, stream>>>(Wk, wkt);
    transpose_w<<<(nW + 255) / 256, 256, 0, stream>>>(Wv, wvt);
    int nO = C_ * C_;
    cvt_f32_bf16<<<(nO + 255) / 256, 256, 0, stream>>>(Wo, wob, nO);
  }

  qkv_gemm<<<dim3(T_ / 32, B_ * H_, 3), 32, 0, stream>>>(xb, wqt, wkt, wvt, qb, kb, vtb);
  flash_attn<<<dim3(T_ / 16, B_ * H_), 32, 0, stream>>>(qb, kb, vtb, obb);
  out_proj<<<dim3((B_ * T_) / 32, C_ / 64), 32, 0, stream>>>(obb, wob, bo, y);
}
